// CNNNet_29454885716038
// MI455X (gfx1250) — matrix operand order repaired and checked by reading, not compiled, not run on hardware
//
#include <hip/hip_runtime.h>

// ---------- types for WMMA ----------
typedef __attribute__((ext_vector_type(16))) __bf16 v16bf;
typedef __attribute__((ext_vector_type(8)))  __bf16 v8bf;
typedef __attribute__((ext_vector_type(8)))  float  v8f;

// ---------- constants ----------
#define VOCAB   32000
#define EMBED   128
#define SEQLEN  2048
#define BATCH   256
#define NFILT   18      // widths 2..19
#define NCOLS   189     // sum of widths
#define NCOLP   192     // padded to 12 N-tiles
#define TPOS    64      // output positions per workgroup
#define SPOS    96      // score positions computed (64 + 18 halo, padded to 6 M-tiles)
#define MT      6
#define NT      12
#define EA_STR  136     // embA row stride in halfwords (272 B, bank-conflict pad)
#define SC_STR  104     // scoresT column stride in halfwords (208 B = 13*16 B, conflict-free)
#define ENC_NEG_INF 0x007FFFFFu   // encf(-inf)

// ---------- helpers ----------
__device__ __forceinline__ unsigned short f2bf(float f) {
    unsigned u = __float_as_uint(f);
    u += 0x7FFFu + ((u >> 16) & 1u);           // round-to-nearest-even
    return (unsigned short)(u >> 16);
}
__device__ __forceinline__ unsigned pk2bf(float lo, float hi) {
    return (unsigned)f2bf(lo) | ((unsigned)f2bf(hi) << 16);
}
__device__ __forceinline__ float bf2f(unsigned short h) {
    return __uint_as_float(((unsigned)h) << 16);
}
// order-preserving float<->uint encoding so unsigned atomicMax == float max
__device__ __forceinline__ unsigned encf(float f) {
    unsigned u = __float_as_uint(f);
    return (u & 0x80000000u) ? ~u : (u | 0x80000000u);
}
__device__ __forceinline__ float decf(unsigned e) {
    unsigned b = (e & 0x80000000u) ? (e & 0x7FFFFFFFu) : ~e;
    return __uint_as_float(b);
}

struct WPtrs { const float* p[NFILT]; };

// ---------- kernel 1: weight repack (f32 -> bf16 columns) + feat init ----------
__global__ void prep_kernel(WPtrs wp, unsigned short* __restrict__ wb,
                            unsigned* __restrict__ featEnc) {
    const int tid = blockIdx.x * 256 + threadIdx.x;
    if (tid < NCOLP * EMBED) {
        const int j = tid >> 7;          // padded column index
        const int k = tid & 127;         // embed index
        unsigned short v = 0;
        if (j < NCOLS) {
            int fi = 0, off = 0;
            for (int i = 0; i < NFILT; ++i) {
                const int o = ((i + 1) * (i + 2)) / 2 - 1;   // column offset of filter i
                if (j < o + (i + 2)) { fi = i; off = o; break; }
            }
            const int dt = j - off;
            v = f2bf(wp.p[fi][dt * EMBED + k]);
        }
        wb[tid] = v;
    }
    if (tid < BATCH * NFILT) featEnc[tid] = ENC_NEG_INF;
}

// ---------- kernel 2: gather + WMMA GEMM + diagonal shift-add + max ----------
__global__ __launch_bounds__(256)
void conv_gemm_kernel(const int* __restrict__ x,
                      const float* __restrict__ table,
                      const unsigned short* __restrict__ wb,
                      unsigned* __restrict__ featEnc) {
    __shared__ __align__(16) unsigned short embA [SPOS  * EA_STR]; // 26112 B bf16 emb rows
    __shared__ __align__(16) unsigned short scrT [NCOLS * SC_STR]; // 39312 B bf16 scores [col][pos]
    __shared__ unsigned lmax[NFILT];

    const int b    = blockIdx.x >> 5;          // 256 batch rows
    const int t0   = (blockIdx.x & 31) * TPOS; // 32 chunks of 64 positions
    const int tid  = threadIdx.x;
    const int wave = tid >> 5;
    const int lane = tid & 31;

    if (tid < NFILT) lmax[tid] = ENC_NEG_INF;

    // ---- phase 1: gather embedding rows -> LDS (bf16), coalesced 512B/row ----
    for (int r = wave; r < SPOS; r += 8) {
        const int t = t0 + r;
        int tok = (t < SEQLEN) ? x[b * SEQLEN + t] : VOCAB;
        if ((unsigned)tok >= (unsigned)(VOCAB + 1)) tok = VOCAB;
        unsigned short h0 = 0, h1 = 0, h2 = 0, h3 = 0;
        if (tok != VOCAB) {                    // padding row == zeros
            const float4 f = *(const float4*)(table + (size_t)tok * EMBED + lane * 4);
            h0 = f2bf(f.x); h1 = f2bf(f.y); h2 = f2bf(f.z); h3 = f2bf(f.w);
        }
        *(ushort4*)(&embA[r * EA_STR + lane * 4]) = make_ushort4(h0, h1, h2, h3);
    }
    __syncthreads();

    // ---- phase 2: scoresT[192 x 96] = (embA[96 x 128] * wb^T)^T via bf16 WMMA ----
    const int lhalf = lane >> 4;   // 0/1: K-half selector
    const int lrow  = lane & 15;   // A row / B column within tile
    for (int task = wave; task < MT * NT; task += 8) {
        const int mt = task / NT;
        const int nt = task % NT;
        // A 16x32 bf16 fragment source: lane<16 -> K {0..7,16..23}, lane>=16 -> +8
        const unsigned short* aBase = &embA[(mt * 16 + lrow) * EA_STR + lhalf * 8];
        // B 32x16 bf16 fragment source: column = lrow, lane<16 -> K {0..7,16..23}, lane>=16 -> +8
        const unsigned short* bBase = wb + (nt * 16 + lrow) * EMBED + lhalf * 8;

        v8f c = {0.f, 0.f, 0.f, 0.f, 0.f, 0.f, 0.f, 0.f};
        #pragma unroll
        for (int k0 = 0; k0 < 4; ++k0) {       // K = 128 = 4 x 32
            v16bf a, bm;
            *((v8bf*)&a)        = *(const v8bf*)(aBase + k0 * 32);
            *(((v8bf*)&a) + 1)  = *(const v8bf*)(aBase + k0 * 32 + 16);
            *((v8bf*)&bm)       = *(const v8bf*)(bBase + k0 * 32);
            *(((v8bf*)&bm) + 1) = *(const v8bf*)(bBase + k0 * 32 + 16);
            c = __builtin_amdgcn_wmma_f32_16x16x32_bf16(
                    false, a, false, bm, (short)0, c, false, false);
        }
        // D 16x16 f32: vgpr r -> position mt*16 + lhalf*8 + r ; column nt*16 + lrow.
        // Column-major score buffer => lane's 8 values are contiguous: pack to one b128 store.
        const int n = nt * 16 + lrow;
        if (n < NCOLS) {
            uint4 pk = make_uint4(pk2bf(c[0], c[1]), pk2bf(c[2], c[3]),
                                  pk2bf(c[4], c[5]), pk2bf(c[6], c[7]));
            *(uint4*)(&scrT[n * SC_STR + mt * 16 + lhalf * 8]) = pk;
        }
    }
    __syncthreads();

    // ---- phase 3: out_fs[t] = sum_dt scoresT[off+dt][t+dt]; max over t ----
    for (int p = tid; p < NFILT * TPOS; p += 256) {
        const int i  = p >> 6;
        const int t  = p & (TPOS - 1);
        const int fs = i + 2;
        const int off = ((i + 1) * (i + 2)) / 2 - 1;
        if (t0 + t + fs <= SEQLEN) {           // valid window only
            float s = 0.f;
            const unsigned short* sp = &scrT[off * SC_STR + t];
            for (int dt = 0; dt < fs; ++dt) s += bf2f(sp[dt * (SC_STR + 1)]);
            atomicMax(&lmax[i], encf(s));
        }
    }
    __syncthreads();
    if (tid < NFILT) atomicMax(&featEnc[b * NFILT + tid], lmax[tid]);
}

// ---------- kernel 3: decode maxes, add bias, 18->2 linear ----------
__global__ void finalize_kernel(const unsigned* __restrict__ featEnc,
                                const float* __restrict__ bias,
                                const float* __restrict__ lw,   // (2,18) row-major
                                const float* __restrict__ lb,
                                float* __restrict__ out) {
    const int b = threadIdx.x;                 // 256 threads, 1 block
    float a0 = lb[0], a1 = lb[1];
    #pragma unroll
    for (int i = 0; i < NFILT; ++i) {
        const float f = decf(featEnc[b * NFILT + i]) + bias[i];
        a0 += f * lw[i];
        a1 += f * lw[NFILT + i];
    }
    out[b * 2 + 0] = a0;
    out[b * 2 + 1] = a1;
}

// ---------- host ----------
extern "C" void kernel_launch(void* const* d_in, const int* in_sizes, int n_in,
                              void* d_out, int out_size, void* d_ws, size_t ws_size,
                              hipStream_t stream) {
    const int*   x     = (const int*)d_in[0];
    const float* table = (const float*)d_in[1];
    WPtrs wp;
    const float *bias, *lw, *lb;
    if (n_in >= 5 + NFILT) {                   // conv_ws tuple flattened to 18 inputs
        for (int i = 0; i < NFILT; ++i) wp.p[i] = (const float*)d_in[2 + i];
        bias = (const float*)d_in[2 + NFILT];
        lw   = (const float*)d_in[3 + NFILT];
        lb   = (const float*)d_in[4 + NFILT];
    } else {                                   // conv_ws passed as one concatenated buffer
        const float* base = (const float*)d_in[2];
        int off = 0;
        for (int i = 0; i < NFILT; ++i) { wp.p[i] = base + off * EMBED; off += i + 2; }
        bias = (const float*)d_in[3];
        lw   = (const float*)d_in[4];
        lb   = (const float*)d_in[5];
    }

    unsigned short* wb      = (unsigned short*)d_ws;                        // 192*128*2 = 49152 B
    unsigned*       featEnc = (unsigned*)((char*)d_ws + NCOLP * EMBED * 2); // 256*18*4 = 18432 B

    prep_kernel<<<(NCOLP * EMBED + 255) / 256, 256, 0, stream>>>(wp, wb, featEnc);
    conv_gemm_kernel<<<BATCH * (SEQLEN / TPOS), 256, 0, stream>>>(x, table, wb, featEnc);
    finalize_kernel<<<1, 256, 0, stream>>>(featEnc, bias, lw, lb, (float*)d_out);
}
